// SimpleLstmCritic_11957188952415
// MI455X (gfx1250) — matrix operand order repaired and checked by reading, not compiled, not run on hardware
//
#include <hip/hip_runtime.h>
#include <hip/hip_bf16.h>
#include <cstdint>

// ----------------------------------------------------------------------------
// SimpleLstmCritic for MI455X (gfx1250), wave32 + WMMA f16 + async-LDS loads.
//
// Pipeline:
//   prep_kernel : pack f32 weights into WMMA B-operand f16 tile layout,
//                 precompute b_ih+b_hh.
//   mlp_kernel  : z = L3(tanh(L2(tanh(L1(x)))))  for all T*B rows,
//                 f16 WMMA 16x16x32, A-tiles hoisted (loaded once per layer
//                 per wave), first WMMA takes inline C=0, bias in epilogue.
//                 z written f16 to workspace (128 MB).
//   lstm_kernel : serial scan over T=512; 128 WGs each own 16 batch rows;
//                 3 WMMAs/step, per-wave weight tiles register-resident,
//                 done-mask folded into the h A-operand scale, value head via
//                 shfl_xor reduction. z[t+1] double-buffered into LDS with
//                 GLOBAL_LOAD_ASYNC_TO_LDS_B128 (ASYNCcnt) + s_wait_asynccnt,
//                 z[t+2] prefetch hint.
//
// Workspace layout (requires ws_size >= 98304 + T*B*64*2 = ~128.1 MB):
//   [0,      90112)  packed f16 weight tiles (W1,W2,W3,W_ih,W_hh)
//   [90112,  90624)  f32 gate-bias sums b_ih+b_hh [128]
//   [98304,  ...  )  f16 z activations [T][B][64]
// ----------------------------------------------------------------------------

typedef __attribute__((ext_vector_type(16))) _Float16 v16h;
typedef __attribute__((ext_vector_type(8)))  _Float16 v8h;
typedef __attribute__((ext_vector_type(4)))  _Float16 v4h;
typedef __attribute__((ext_vector_type(8)))  float    v8f;

#define N_OBS    64
#define N_H1     128
#define N_H2     128
#define N_LIN    64
#define N_LH     32
#define TT       512
#define BB       2048

// packed weight tile offsets (in halfs)
#define OFF_W1   0        // 8 ntiles * 2 ktiles * 512
#define OFF_W2   8192     // 8 * 4 * 512
#define OFF_W3   24576    // 4 * 4 * 512
#define OFF_WIH  32768    // 8 * 2 * 512
#define OFF_WHH  40960    // 8 * 1 * 512
#define PACK_TOTAL 45056

union HV16 { v16h v; struct { v8h lo, hi; } p; };

// A operand (16x32 f16) from an LDS row-major tile.
// lane L -> row M=L&15; halves 0-7 = K[(L>>4)*8 .. +7],
// halves 8-15 = K[16 + (L>>4)*8 .. +7]  ->  two 16B LDS loads per lane.
__device__ __forceinline__ v16h load_a(const _Float16* s, int stride, int r0,
                                       int k0, int lane) {
  const int M = lane & 15, g = (lane >> 4) * 8;
  const _Float16* p = s + (r0 + M) * stride + k0 + g;
  HV16 u;
  u.p.lo = *(const v8h*)(p);
  u.p.hi = *(const v8h*)(p + 16);
  return u.v;
}

// B operand (32x16 f16) from a pre-packed global tile (512 halfs):
// lane L holds col n=L&15, K = (s>>3)*16 + (L>>4)*8 + (s&7)  (16 halfs per lane).
__device__ __forceinline__ v16h load_b(const _Float16* gptr, int lane) {
  const _Float16* p = gptr + lane * 16;
  HV16 u;
  u.p.lo = *(const v8h*)(p);
  u.p.hi = *(const v8h*)(p + 8);
  return u.v;
}

// CDNA5 async global->LDS copy (16 bytes per lane), tracked by ASYNCcnt.
__device__ __forceinline__ void async_ld_b128(unsigned lds_addr, const void* gaddr) {
  asm volatile("global_load_async_to_lds_b128 %0, %1, off"
               :: "v"(lds_addr), "v"(gaddr)
               : "memory");
}
__device__ __forceinline__ void wait_asynccnt0() {
  asm volatile("s_wait_asynccnt 0x0" ::: "memory");
}

__device__ __forceinline__ float fexp2(float x) { return __builtin_amdgcn_exp2f(x); }
__device__ __forceinline__ float frcp(float x)  { return __builtin_amdgcn_rcpf(x); }
__device__ __forceinline__ float fsig(float x)  { return frcp(1.0f + fexp2(-1.44269504f * x)); }
__device__ __forceinline__ float ftanh(float x) { return 1.0f - 2.0f * frcp(fexp2(2.88539008f * x) + 1.0f); }

// ---------------------------------------------------------------------------
// Weight packing: out[tile][lane*16 + s] = W[n][k] per load_b layout.
// ---------------------------------------------------------------------------
__global__ void prep_kernel(const float* __restrict__ W1, const float* __restrict__ W2,
                            const float* __restrict__ W3, const float* __restrict__ Wih,
                            const float* __restrict__ Whh, const float* __restrict__ bih,
                            const float* __restrict__ bhh,
                            _Float16* __restrict__ wp, float* __restrict__ bg) {
  for (int i = blockIdx.x * blockDim.x + threadIdx.x; i < PACK_TOTAL + 128;
       i += gridDim.x * blockDim.x) {
    if (i < PACK_TOTAL) {
      const float* W; int K, ktiles, base, idx = i;
      if (idx < OFF_W2)       { W = W1;  K = 64;  ktiles = 2; base = OFF_W1; }
      else if (idx < OFF_W3)  { W = W2;  K = 128; ktiles = 4; base = OFF_W2;  idx -= OFF_W2; }
      else if (idx < OFF_WIH) { W = W3;  K = 128; ktiles = 4; base = OFF_W3;  idx -= OFF_W3; }
      else if (idx < OFF_WHH) { W = Wih; K = 64;  ktiles = 2; base = OFF_WIH; idx -= OFF_WIH; }
      else                    { W = Whh; K = 32;  ktiles = 1; base = OFF_WHH; idx -= OFF_WHH; }
      const int tile = idx >> 9, r = idx & 511;
      const int L = r >> 4, s = r & 15;
      const int nt = tile / ktiles, kt = tile - nt * ktiles;
      const int n = nt * 16 + (L & 15);
      const int k = kt * 32 + ((s >> 3) << 4) + ((L >> 4) << 3) + (s & 7);
      wp[base + idx] = (_Float16)W[n * K + k];
    } else {
      const int j = i - PACK_TOTAL;
      bg[j] = bih[j] + bhh[j];
    }
  }
}

// ---------------------------------------------------------------------------
// MLP encoder: 64 rows per WG, 8 waves = (4 row-tiles) x (2 N-halves).
// ---------------------------------------------------------------------------
__global__ __launch_bounds__(256) void mlp_kernel(
    const float* __restrict__ x, const _Float16* __restrict__ wp,
    const float* __restrict__ b1, const float* __restrict__ b2,
    const float* __restrict__ b3, _Float16* __restrict__ z) {
  __shared__ __attribute__((aligned(16))) _Float16 sX[64 * 72];   // x (f16) / z3 staging
  __shared__ __attribute__((aligned(16))) _Float16 s1[64 * 136];  // layer-1 activations
  __shared__ __attribute__((aligned(16))) _Float16 s2[64 * 136];  // layer-2 activations

  const int tid = threadIdx.x, lane = tid & 31, wid = tid >> 5;
  const int m = wid & 3, nh = wid >> 2;
  const int r0 = m * 16;
  const long rows0 = (long)blockIdx.x * 64;
  const _Float16* W1p = wp + OFF_W1;
  const _Float16* W2p = wp + OFF_W2;
  const _Float16* W3p = wp + OFF_W3;

  // stage x tile (64 rows x 64 f32), convert to f16 in LDS
#pragma unroll
  for (int j = 0; j < 4; ++j) {
    const int f4 = tid + j * 256;          // 1024 float4
    const int row = f4 >> 4, c4 = f4 & 15;
    const float4 v = *(const float4*)(x + (rows0 + row) * N_OBS + c4 * 4);
    v4h h;
    h.x = (_Float16)v.x; h.y = (_Float16)v.y;
    h.z = (_Float16)v.z; h.w = (_Float16)v.w;
    *(v4h*)(&sX[row * 72 + c4 * 4]) = h;
  }
  __syncthreads();

  const int M0 = (lane >> 4) * 8, N = lane & 15;

  // layer 1: [64x64] x [64x128] -> tanh -> s1  (A tiles hoisted)
  {
    v16h aa[2];
#pragma unroll
    for (int kt = 0; kt < 2; ++kt) aa[kt] = load_a(sX, 72, r0, kt * 32, lane);
#pragma unroll
    for (int i = 0; i < 4; ++i) {
      const int nt = nh * 4 + i;
      const float b = b1[nt * 16 + N];
      v8f acc = {};
#pragma unroll
      for (int kt = 0; kt < 2; ++kt) {
        v16h w = load_b(W1p + (nt * 2 + kt) * 512, lane);
        acc = __builtin_amdgcn_wmma_f32_16x16x32_f16(false, aa[kt], false, w, (short)0, acc, false, false);
      }
#pragma unroll
      for (int v = 0; v < 8; ++v)
        s1[(r0 + M0 + v) * 136 + nt * 16 + N] = (_Float16)ftanh(acc[v] + b);
    }
  }
  __syncthreads();

  // layer 2: [64x128] x [128x128] -> tanh -> s2
  {
    v16h aa[4];
#pragma unroll
    for (int kt = 0; kt < 4; ++kt) aa[kt] = load_a(s1, 136, r0, kt * 32, lane);
#pragma unroll
    for (int i = 0; i < 4; ++i) {
      const int nt = nh * 4 + i;
      const float b = b2[nt * 16 + N];
      v8f acc = {};
#pragma unroll
      for (int kt = 0; kt < 4; ++kt) {
        v16h w = load_b(W2p + (nt * 4 + kt) * 512, lane);
        acc = __builtin_amdgcn_wmma_f32_16x16x32_f16(false, aa[kt], false, w, (short)0, acc, false, false);
      }
#pragma unroll
      for (int v = 0; v < 8; ++v)
        s2[(r0 + M0 + v) * 136 + nt * 16 + N] = (_Float16)ftanh(acc[v] + b);
    }
  }
  __syncthreads();

  // layer 3: [64x128] x [128x64] -> f16 -> sX (reused, stride 64)
  {
    v16h aa[4];
#pragma unroll
    for (int kt = 0; kt < 4; ++kt) aa[kt] = load_a(s2, 136, r0, kt * 32, lane);
#pragma unroll
    for (int i = 0; i < 2; ++i) {
      const int nt = nh * 2 + i;
      const float b = b3[nt * 16 + N];
      v8f acc = {};
#pragma unroll
      for (int kt = 0; kt < 4; ++kt) {
        v16h w = load_b(W3p + (nt * 4 + kt) * 512, lane);
        acc = __builtin_amdgcn_wmma_f32_16x16x32_f16(false, aa[kt], false, w, (short)0, acc, false, false);
      }
#pragma unroll
      for (int v = 0; v < 8; ++v)
        sX[(r0 + M0 + v) * 64 + nt * 16 + N] = (_Float16)(acc[v] + b);
    }
  }
  __syncthreads();

  // coalesced f16 store of z tile (64 x 64)
#pragma unroll
  for (int j = 0; j < 2; ++j) {
    const int idx = tid + j * 256;         // 512 x 16B
    const uint4 v = *(const uint4*)(&sX[idx * 8]);
    *(uint4*)(z + rows0 * 64 + idx * 8) = v;
  }
}

// ---------------------------------------------------------------------------
// Masked LSTM scan + value head. One WG = 16 batch rows for all T steps.
// z_t double-buffered in LDS via async global->LDS loads (ASYNCcnt).
// ---------------------------------------------------------------------------
__global__ __launch_bounds__(256) void lstm_kernel(
    const _Float16* __restrict__ z, const float* __restrict__ done,
    const float* __restrict__ h0, const float* __restrict__ c0,
    const _Float16* __restrict__ wp, const float* __restrict__ bg,
    const float* __restrict__ Wv, const float* __restrict__ bv,
    float* __restrict__ out) {
  __shared__ __attribute__((aligned(16))) _Float16 sz[2][16 * 72];  // z_t double buffer
  __shared__ __attribute__((aligned(16))) _Float16 sh[16 * 40];     // h (f16)
  __shared__ __attribute__((aligned(16))) float    sc[16 * 32];     // c (f32)
  __shared__ __attribute__((aligned(16))) float    sg[16 * 132];    // gates
  __shared__ float smask[16];

  const int tid = threadIdx.x, lane = tid & 31, wid = tid >> 5;
  const int b0 = blockIdx.x * 16;
  const _Float16* Wihp = wp + OFF_WIH;
  const _Float16* Whhp = wp + OFF_WHH;

  // per-wave register-resident weight tiles for the whole scan
  const v16h Bih0 = load_b(Wihp + (wid * 2 + 0) * 512, lane);
  const v16h Bih1 = load_b(Wihp + (wid * 2 + 1) * 512, lane);
  const v16h Bhh  = load_b(Whhp + wid * 512, lane);
  const float gb  = bg[wid * 16 + (lane & 15)];

  const int q = tid & 15, r = tid >> 4;
  const float wv0 = Wv[q], wv1 = Wv[q + 16], bvv = bv[0];

  // init h, c
#pragma unroll
  for (int k = 0; k < 2; ++k) {
    const int e = tid + k * 256, rr = e >> 5, jj = e & 31;
    sc[rr * 32 + jj] = c0[(b0 + rr) * N_LH + jj];
    sh[rr * 40 + jj] = (_Float16)h0[(b0 + rr) * N_LH + jj];
  }
  // async preload of z_0 (16 rows x 64 f16 = 128 lanes x 16B)
  if (tid < 128) {
    const int row = tid >> 3, c8 = tid & 7;
    async_ld_b128((unsigned)(uintptr_t)&sz[0][row * 72 + c8 * 8],
                  z + ((long)0 * BB + b0 + row) * 64 + c8 * 8);
  }
  wait_asynccnt0();
  __syncthreads();

  for (int t = 0; t < TT; ++t) {
    const int cur = t & 1, nxt = cur ^ 1;
    if (tid < 16) smask[tid] = 1.0f - done[(long)t * BB + b0 + tid];
    __syncthreads();  // (A) sz[cur] complete; sz[nxt] free; smask/h/c ready

    // kick async copy of z[t+1] into the alternate LDS buffer
    if ((t + 1 < TT) && (tid < 128)) {
      const int row = tid >> 3, c8 = tid & 7;
      async_ld_b128((unsigned)(uintptr_t)&sz[nxt][row * 72 + c8 * 8],
                    z + ((long)(t + 1) * BB + b0 + row) * 64 + c8 * 8);
    }
    if ((t + 2 < TT) && (tid < 16))
      __builtin_prefetch(z + ((long)(t + 2) * BB + b0 + tid) * 64, 0, 1);

    // gates = z_t @ W_ih.T + (mask*h) @ W_hh.T + (b_ih + b_hh)
    v8f acc = {};
    v16h az0 = load_a(sz[cur], 72, 0, 0, lane);
    v16h az1 = load_a(sz[cur], 72, 0, 32, lane);
    acc = __builtin_amdgcn_wmma_f32_16x16x32_f16(false, az0, false, Bih0, (short)0, acc, false, false);
    acc = __builtin_amdgcn_wmma_f32_16x16x32_f16(false, az1, false, Bih1, (short)0, acc, false, false);
    v16h ah = load_a(sh, 40, 0, 0, lane);
    const _Float16 mh = (_Float16)smask[lane & 15];  // mask is per-row (M = lane&15)
#pragma unroll
    for (int i = 0; i < 16; ++i) ah[i] *= mh;
    acc = __builtin_amdgcn_wmma_f32_16x16x32_f16(false, ah, false, Bhh, (short)0, acc, false, false);
    {
      const int M0 = (lane >> 4) * 8, N = lane & 15;
#pragma unroll
      for (int v = 0; v < 8; ++v) sg[(M0 + v) * 132 + wid * 16 + N] = acc[v] + gb;
    }
    __syncthreads();  // (B) gates visible

    // elementwise LSTM update + value head (PyTorch gate order i,f,g,o)
    float vp = 0.0f;
#pragma unroll
    for (int k2 = 0; k2 < 2; ++k2) {
      const int j = q + k2 * 16;
      const float gi = sg[r * 132 + j];
      const float gf = sg[r * 132 + 32 + j];
      const float gg = sg[r * 132 + 64 + j];
      const float go = sg[r * 132 + 96 + j];
      float c = sc[r * 32 + j] * smask[r];
      c = fsig(gf) * c + fsig(gi) * ftanh(gg);
      const float h = fsig(go) * ftanh(c);
      sc[r * 32 + j] = c;
      sh[r * 40 + j] = (_Float16)h;
      vp += h * (k2 ? wv1 : wv0);
    }
#pragma unroll
    for (int off = 8; off > 0; off >>= 1) vp += __shfl_xor(vp, off, 16);
    if (q == 0) out[(long)t * BB + b0 + r] = vp + bvv;

    wait_asynccnt0();   // issuing waves: z[t+1] landed in LDS
    __syncthreads();    // (C) publish sz[nxt] + sh/sc to all waves
  }
}

extern "C" void kernel_launch(void* const* d_in, const int* in_sizes, int n_in,
                              void* d_out, int out_size, void* d_ws, size_t ws_size,
                              hipStream_t stream) {
  (void)in_sizes; (void)n_in; (void)out_size; (void)ws_size;
  const float* x    = (const float*)d_in[0];
  const float* done = (const float*)d_in[1];
  const float* h0   = (const float*)d_in[2];
  const float* c0   = (const float*)d_in[3];
  const float* W1   = (const float*)d_in[4];
  const float* b1   = (const float*)d_in[5];
  const float* W2   = (const float*)d_in[6];
  const float* b2   = (const float*)d_in[7];
  const float* W3   = (const float*)d_in[8];
  const float* b3   = (const float*)d_in[9];
  const float* Wih  = (const float*)d_in[10];
  const float* Whh  = (const float*)d_in[11];
  const float* bih  = (const float*)d_in[12];
  const float* bhh  = (const float*)d_in[13];
  const float* Wv   = (const float*)d_in[14];
  const float* bv   = (const float*)d_in[15];

  _Float16* wp = (_Float16*)d_ws;
  float* bg    = (float*)((char*)d_ws + 90112);
  _Float16* z  = (_Float16*)((char*)d_ws + 98304);  // needs ws >= ~128.1 MB
  float* out   = (float*)d_out;

  prep_kernel<<<64, 256, 0, stream>>>(W1, W2, W3, Wih, Whh, bih, bhh, wp, bg);
  mlp_kernel<<<(TT * BB) / 64, 256, 0, stream>>>(x, wp, b1, b2, b3, z);
  lstm_kernel<<<BB / 16, 256, 0, stream>>>(z, done, h0, c0, wp, bg, Wv, bv, out);
}
